// TwoAttentionLayer_47132971107261
// MI455X (gfx1250) — compile-verified, matrix operand order checked by reading, not run on hardware
//
#include <hip/hip_runtime.h>
#include <stdint.h>

typedef __attribute__((ext_vector_type(16))) _Float16 v16h;
typedef __attribute__((ext_vector_type(8)))  float    v8f;
typedef unsigned int __attribute__((ext_vector_type(4))) v4u;
typedef int __attribute__((ext_vector_type(4))) v4i;
typedef int __attribute__((ext_vector_type(8))) v8i;

#define B_   4
#define L_   1024
#define D_   512
#define H_   8
#define E_   64
#define F_   64
#define DFF  2048
#define BL   (B_ * L_)          // 4096
#define BLD  (B_ * L_ * D_)     // 2097152

// ---------------------------------------------------------------------------
// f32 -> f16 conversion
// ---------------------------------------------------------------------------
__global__ void f32_to_f16_kernel(const float* __restrict__ x, _Float16* __restrict__ y, int n) {
  int i = blockIdx.x * blockDim.x + threadIdx.x;
  if (i < n) y[i] = (_Float16)x[i];
}

__global__ void zero_f32_kernel(float* __restrict__ p, int n) {
  int i = blockIdx.x * blockDim.x + threadIdx.x;
  if (i < n) p[i] = 0.0f;
}

__global__ void add_kernel(const float* __restrict__ a, const float* __restrict__ b,
                           float* __restrict__ c, int n) {
  int i = blockIdx.x * blockDim.x + threadIdx.x;
  if (i < n) c[i] = a[i] + b[i];
}

// router [F,D] f32 -> replicated [B,F,D] f16
__global__ void router_rep_kernel(const float* __restrict__ r, _Float16* __restrict__ out) {
  int i = blockIdx.x * blockDim.x + threadIdx.x;
  if (i < B_ * F_ * D_) out[i] = (_Float16)r[i & (F_ * D_ - 1)];
}

// ---------------------------------------------------------------------------
// WMMA fragment load (CDNA5 ISA 7.12.2 16-bit striping)
// ---------------------------------------------------------------------------
#define KC 32

__device__ __forceinline__ v16h load_frag(const _Float16* tile, int row, int half) {
  // tile: row-major [rows][KC] f16 in LDS.
  // VGPR v (v<4): K = 2v+8*half, 2v+1+8*half ; VGPR 4+v: K = 16+2v+8*half, ...
  const uint32_t* p = (const uint32_t*)(tile + row * KC);  // 16 dwords/row
  union { v16h v; uint32_t d[8]; } u;
  int o = 4 * half;
#pragma unroll
  for (int j = 0; j < 4; ++j) u.d[j]     = p[o + j];
#pragma unroll
  for (int j = 0; j < 4; ++j) u.d[4 + j] = p[8 + o + j];
  return u.v;
}

// ---------------------------------------------------------------------------
// WMMA GEMM (vector-load path):
//   C[M,N] = A[M,K] * W[N,K]^T (+bias)(+add)(relu)
// 256 threads = 8 waves; block tile 64x64; wave tile 16x32 (2 WMMAs, A reuse).
// ---------------------------------------------------------------------------
__global__ __launch_bounds__(256) void gemm_f16_wmma(
    const _Float16* __restrict__ A,   // [M,K]
    const _Float16* __restrict__ Bw,  // [N,K]  (torch Linear weight layout)
    const float* __restrict__ bias,   // [N] or nullptr
    const float* __restrict__ add,    // [M,N] or nullptr
    float* __restrict__ C,            // [M,N]
    int M, int N, int K, int relu) {
  __shared__ __align__(16) _Float16 As[64 * KC];
  __shared__ __align__(16) _Float16 Bs[64 * KC];

  const int tid   = threadIdx.x;
  const int lane  = tid & 31;
  const int wave  = tid >> 5;
  const int waveM = wave & 3;
  const int waveN = wave >> 2;
  const int bm = blockIdx.y * 64;
  const int bn = blockIdx.x * 64;

  v8f acc0 = {0.f, 0.f, 0.f, 0.f, 0.f, 0.f, 0.f, 0.f};
  v8f acc1 = {0.f, 0.f, 0.f, 0.f, 0.f, 0.f, 0.f, 0.f};

  const int ldRow = tid >> 2;        // 0..63
  const int ldCol = (tid & 3) * 8;   // 0,8,16,24 (f16 units; 16B granules)
  const int half = lane >> 4;
  const int rl   = lane & 15;

  for (int k0 = 0; k0 < K; k0 += KC) {
    if (k0 + KC < K) {  // pull next K-chunk toward caches (global_prefetch_b8)
      __builtin_prefetch(A  + (size_t)(bm + ldRow) * K + k0 + KC + ldCol, 0, 1);
      __builtin_prefetch(Bw + (size_t)(bn + ldRow) * K + k0 + KC + ldCol, 0, 1);
    }
    *(uint4*)(&As[ldRow * KC + ldCol]) = *(const uint4*)(A  + (size_t)(bm + ldRow) * K + k0 + ldCol);
    *(uint4*)(&Bs[ldRow * KC + ldCol]) = *(const uint4*)(Bw + (size_t)(bn + ldRow) * K + k0 + ldCol);
    __syncthreads();

    v16h af = load_frag(As, 16 * waveM + rl, half);
    v16h b0 = load_frag(Bs, 32 * waveN + rl, half);
    v16h b1 = load_frag(Bs, 32 * waveN + 16 + rl, half);
    acc0 = __builtin_amdgcn_wmma_f32_16x16x32_f16(false, af, false, b0, (short)0, acc0, false, false);
    acc1 = __builtin_amdgcn_wmma_f32_16x16x32_f16(false, af, false, b1, (short)0, acc1, false, false);
    __syncthreads();
  }

#pragma unroll
  for (int r = 0; r < 8; ++r) {
    int m  = bm + 16 * waveM + r + 8 * half;
    int n0 = bn + 32 * waveN + rl;
    int n1 = n0 + 16;
    float v0 = acc0[r], v1 = acc1[r];
    if (bias) { v0 += bias[n0]; v1 += bias[n1]; }
    if (add)  { v0 += add[(size_t)m * N + n0]; v1 += add[(size_t)m * N + n1]; }
    if (relu) { v0 = fmaxf(v0, 0.f); v1 = fmaxf(v1, 0.f); }
    C[(size_t)m * N + n0] = v0;
    C[(size_t)m * N + n1] = v1;
  }
}

// ---------------------------------------------------------------------------
// Tensor Data Mover tile fetch: 64 rows x 32 f16 (2D tile) global -> LDS.
// D# built per CDNA5 ISA §8.3 (Group0) / §8.4 (Group1).
// ---------------------------------------------------------------------------
__device__ __forceinline__ void tdm_load_tile_64x32(const _Float16* gbase,
                                                    int rowStrideElems,
                                                    uint32_t lds_off) {
  uint64_t ga = (uint64_t)(uintptr_t)gbase;
  v4u g0;
  g0[0] = 1u;                                               // count=1, user mode
  g0[1] = lds_off;                                          // lds_addr (bytes)
  g0[2] = (uint32_t)ga;                                     // global_addr[31:0]
  g0[3] = (uint32_t)((ga >> 32) & 0x01FFFFFFu) | (2u << 30);// global_addr[56:32] | type=2
  v8i g1;
  g1[0] = 0x10000;          // workgroup_mask=0, data_size=1 (2 bytes)
  g1[1] = (KC << 16);       // atomic_barrier_addr=0 | tensor_dim0[15:0]=32
  g1[2] = (64 << 16);       // tensor_dim0[31:16]=0  | tensor_dim1[15:0]=64
  g1[3] = (KC << 16);       // tensor_dim1[31:16]=0  | tile_dim0=32
  g1[4] = 64;               // tile_dim1=64 | tile_dim2=0
  g1[5] = rowStrideElems;   // tensor_dim0_stride[31:0] (data_size units)
  g1[6] = 0;                // stride hi | tensor_dim1_stride lo
  g1[7] = 0;
  v4i z4 = {0, 0, 0, 0};
#if __clang_major__ >= 23
  v8i z8 = {0, 0, 0, 0, 0, 0, 0, 0};
  __builtin_amdgcn_tensor_load_to_lds(g0, g1, z4, z4, z8, 0);
#else
  __builtin_amdgcn_tensor_load_to_lds(g0, g1, z4, z4, 0);
#endif
}

// ---------------------------------------------------------------------------
// WMMA GEMM (TDM path): A and W tiles streamed to LDS by the Tensor Data
// Mover (wave 0 issues descriptors, waits on TENSORcnt, barrier releases all).
// Used for the large MLP GEMMs.
// ---------------------------------------------------------------------------
__global__ __launch_bounds__(256) void gemm_f16_wmma_tdm(
    const _Float16* __restrict__ A, const _Float16* __restrict__ Bw,
    const float* __restrict__ bias, const float* __restrict__ add,
    float* __restrict__ C, int M, int N, int K, int relu) {
  __shared__ __align__(16) _Float16 As[64 * KC];
  __shared__ __align__(16) _Float16 Bs[64 * KC];

  const int tid  = threadIdx.x;
  const int lane = tid & 31;
  const int wave = tid >> 5;
  const int waveM = wave & 3;
  const int waveN = wave >> 2;
  const int bm = blockIdx.y * 64;
  const int bn = blockIdx.x * 64;
  const int half = lane >> 4;
  const int rl   = lane & 15;

  const uint32_t ldsA = (uint32_t)(uintptr_t)&As[0];
  const uint32_t ldsB = (uint32_t)(uintptr_t)&Bs[0];

  v8f acc0 = {0.f, 0.f, 0.f, 0.f, 0.f, 0.f, 0.f, 0.f};
  v8f acc1 = {0.f, 0.f, 0.f, 0.f, 0.f, 0.f, 0.f, 0.f};

  for (int k0 = 0; k0 < K; k0 += KC) {
    if (wave == 0) {
      tdm_load_tile_64x32(A  + (size_t)bm * K + k0, K, ldsA);
      tdm_load_tile_64x32(Bw + (size_t)bn * K + k0, K, ldsB);
      __builtin_amdgcn_s_wait_tensorcnt(0);   // s_wait_tensorcnt 0
    }
    __syncthreads();

    v16h af = load_frag(As, 16 * waveM + rl, half);
    v16h b0 = load_frag(Bs, 32 * waveN + rl, half);
    v16h b1 = load_frag(Bs, 32 * waveN + 16 + rl, half);
    acc0 = __builtin_amdgcn_wmma_f32_16x16x32_f16(false, af, false, b0, (short)0, acc0, false, false);
    acc1 = __builtin_amdgcn_wmma_f32_16x16x32_f16(false, af, false, b1, (short)0, acc1, false, false);
    __syncthreads();
  }

#pragma unroll
  for (int r = 0; r < 8; ++r) {
    int m  = bm + 16 * waveM + r + 8 * half;
    int n0 = bn + 32 * waveN + rl;
    int n1 = n0 + 16;
    float v0 = acc0[r], v1 = acc1[r];
    if (bias) { v0 += bias[n0]; v1 += bias[n1]; }
    if (add)  { v0 += add[(size_t)m * N + n0]; v1 += add[(size_t)m * N + n1]; }
    if (relu) { v0 = fmaxf(v0, 0.f); v1 = fmaxf(v1, 0.f); }
    C[(size_t)m * N + n0] = v0;
    C[(size_t)m * N + n1] = v1;
  }
}

// ---------------------------------------------------------------------------
// Fused trend attention: per (b,h,l) row, softmax over s, atomic col-sum.
// ---------------------------------------------------------------------------
__global__ __launch_bounds__(256) void trend_attn_colsum(
    const float* __restrict__ Q,  // [B, LqFull, H, E]
    const float* __restrict__ K,  // [B, 1024,  H, E]
    float* __restrict__ wsum,     // [B, H, 1024]
    int stride, int LqFull) {
  const int l = blockIdx.x, h = blockIdx.y, b = blockIdx.z;
  __shared__ float qs[E_];
  __shared__ float row[L_];
  __shared__ float red[256];
  const int tid = threadIdx.x;

  const float* qp = Q + (((size_t)b * LqFull + (size_t)l * stride) * H_ + h) * E_;
  if (tid < E_) qs[tid] = qp[tid];
  __syncthreads();

  float lmax = -3.4e38f;
#pragma unroll
  for (int j = 0; j < 4; ++j) {
    int s = tid + 256 * j;
    const float* kp = K + (((size_t)b * L_ + s) * H_ + h) * E_;
    float acc = 0.f;
#pragma unroll
    for (int e = 0; e < E_; ++e) acc += qs[e] * kp[e];
    row[s] = acc;
    lmax = fmaxf(lmax, acc);
  }
  red[tid] = lmax; __syncthreads();
  for (int off = 128; off > 0; off >>= 1) {
    if (tid < off) red[tid] = fmaxf(red[tid], red[tid + off]);
    __syncthreads();
  }
  float mx = red[0]; __syncthreads();

  float lsum = 0.f;
#pragma unroll
  for (int j = 0; j < 4; ++j) {
    int s = tid + 256 * j;
    float ex = __expf(row[s] - mx);
    row[s] = ex; lsum += ex;
  }
  red[tid] = lsum; __syncthreads();
  for (int off = 128; off > 0; off >>= 1) {
    if (tid < off) red[tid] += red[tid + off];
    __syncthreads();
  }
  float inv = 1.f / red[0];
#pragma unroll
  for (int j = 0; j < 4; ++j) {
    int s = tid + 256 * j;
    atomicAdd(&wsum[((size_t)b * H_ + h) * L_ + s], row[s] * inv);
  }
}

// out[b,s,h*E+e] = V[b,s,h,e] * wsum[b,h,s] * scale
__global__ void trend_apply_kernel(const float* __restrict__ V, const float* __restrict__ wsum,
                                   float* __restrict__ out, float scale) {
  int i = blockIdx.x * blockDim.x + threadIdx.x;
  if (i >= BLD) return;
  int d = i & (D_ - 1);
  int s = (i >> 9) & (L_ - 1);
  int b = i >> 19;
  int h = d >> 6;
  out[i] = V[i] * wsum[((size_t)b * H_ + h) * L_ + s] * scale;
}

// ---------------------------------------------------------------------------
// Season: circular cross-correlation corr[b,h,e,tau] = sum_l q[l]*k[(l+tau)%L]
// ---------------------------------------------------------------------------
__global__ __launch_bounds__(256) void season_corr_kernel(
    const float* __restrict__ Q, const float* __restrict__ K, float* __restrict__ corr) {
  const int e = blockIdx.x, h = blockIdx.y, b = blockIdx.z;
  __shared__ float qs[L_];
  __shared__ float ks[L_];
  const int tid = threadIdx.x;
  for (int l = tid; l < L_; l += 256) {
    qs[l] = Q[(((size_t)b * L_ + l) * H_ + h) * E_ + e];
    ks[l] = K[(((size_t)b * L_ + l) * H_ + h) * E_ + e];
  }
  __syncthreads();
#pragma unroll
  for (int j = 0; j < 4; ++j) {
    int tau = tid + 256 * j;
    float acc = 0.f;
    for (int l = 0; l < L_; ++l) acc += qs[l] * ks[(l + tau) & (L_ - 1)];
    corr[(((size_t)b * H_ + h) * E_ + e) * L_ + tau] = acc;
  }
}

// mv[b,tau] = mean over (h,e) of corr
__global__ void corr_mean_kernel(const float* __restrict__ corr, float* __restrict__ mv) {
  int i = blockIdx.x * blockDim.x + threadIdx.x;
  if (i >= B_ * L_) return;
  int b = i >> 10, tau = i & (L_ - 1);
  const float* base = corr + (size_t)b * (H_ * E_) * L_ + tau;
  float acc = 0.f;
  for (int he = 0; he < H_ * E_; ++he) acc += base[(size_t)he * L_];
  mv[i] = acc * (1.f / (H_ * E_));
}

// top-10 of mean-over-b(mv), then softmax of mv[b, idx] over the 10 picks
__global__ void topk_softmax_kernel(const float* __restrict__ mv,
                                    int* __restrict__ idx, float* __restrict__ smap) {
  __shared__ float g[L_];
  __shared__ int sidx[16];
  const int tid = threadIdx.x;
  for (int t = tid; t < L_; t += 64)
    g[t] = mv[t] + mv[L_ + t] + mv[2 * L_ + t] + mv[3 * L_ + t];
  __syncthreads();
  if (tid == 0) {
    for (int j = 0; j < 10; ++j) {
      int best = 0; float bv = -3.4e38f;
      for (int t = 0; t < L_; ++t) if (g[t] > bv) { bv = g[t]; best = t; }
      sidx[j] = best; g[best] = -3.4e38f;
    }
    for (int j = 0; j < 10; ++j) idx[j] = sidx[j];
    for (int b = 0; b < B_; ++b) {
      float mx = -3.4e38f;
      for (int j = 0; j < 10; ++j) mx = fmaxf(mx, mv[b * L_ + sidx[j]]);
      float ex[10]; float s = 0.f;
      for (int j = 0; j < 10; ++j) { ex[j] = __expf(mv[b * L_ + sidx[j]] - mx); s += ex[j]; }
      for (int j = 0; j < 10; ++j) smap[b * 16 + j] = ex[j] / s;
    }
  }
}

// qsum[i][b,h,e] = sum over strided l of Qtr
__global__ void qsum_kernel(const float* __restrict__ Qtr, float* __restrict__ qsum) {
  int i = blockIdx.x * blockDim.x + threadIdx.x;  // 3*B*H*E = 6144
  if (i >= 3 * B_ * H_ * E_) return;
  int e = i & 63, h = (i >> 6) & 7, b = (i >> 9) & 3, it = i >> 11;
  int stride = it + 1;
  int Lq = (L_ + stride - 1) / stride;
  float acc = 0.f;
  for (int l = 0; l < Lq; ++l)
    acc += Qtr[(((size_t)b * L_ + l * stride) * H_ + h) * E_ + e];
  qsum[i] = acc;
}

// cs[i][b,h,s] = <qsum[i][b,h,:], Ktr[b,s,h,:]>  (== colsum of raw attn)
__global__ void colsum_kernel(const float* __restrict__ qsum, const float* __restrict__ Ktr,
                              float* __restrict__ cs) {
  int i = blockIdx.x * blockDim.x + threadIdx.x;  // 3*B*H*L = 98304
  if (i >= 3 * B_ * H_ * L_) return;
  int s = i & (L_ - 1), h = (i >> 10) & 7, b = (i >> 13) & 3, it = i >> 15;
  const float* q = qsum + (((size_t)it * B_ + b) * H_ + h) * E_;
  const float* k = Ktr + (((size_t)b * L_ + s) * H_ + h) * E_;
  float acc = 0.f;
#pragma unroll
  for (int e = 0; e < E_; ++e) acc += q[e] * k[e];
  cs[i] = acc;
}

// delays (output flat [B,H,E,L] == faithful reshape to [B,L,D])
__global__ void delays_kernel(const float* __restrict__ V, const int* __restrict__ idx,
                              const float* __restrict__ smap, const float* __restrict__ cs,
                              float* __restrict__ out) {
  int i = blockIdx.x * blockDim.x + threadIdx.x;
  if (i >= BLD) return;
  int s = i & (L_ - 1);
  int e = (i >> 10) & 63;
  int h = (i >> 16) & 7;
  int b = i >> 19;
  float acc = 0.f;
#pragma unroll
  for (int it = 0; it < 3; ++it) {
    int es = (e + idx[it]) & (E_ - 1);
    float pat = V[(((size_t)b * L_ + s) * H_ + h) * E_ + es];
    float w = smap[b * 16 + it];
    if (cs) w += cs[(((size_t)it * B_ + b) * H_ + h) * L_ + s];
    acc += pat * w;
  }
  out[i] = acc;
}

// y = LN(alpha*x)*g + b ; one block per row of 512
__global__ __launch_bounds__(256) void layernorm_kernel(
    const float* __restrict__ x, const float* __restrict__ g, const float* __restrict__ bb,
    float* __restrict__ y, float alpha) {
  const int row = blockIdx.x, tid = threadIdx.x;
  __shared__ float red[256];
  const float* xr = x + (size_t)row * D_;
  float v0 = alpha * xr[tid], v1 = alpha * xr[tid + 256];
  red[tid] = v0 + v1; __syncthreads();
  for (int off = 128; off > 0; off >>= 1) { if (tid < off) red[tid] += red[tid + off]; __syncthreads(); }
  float mean = red[0] * (1.f / D_); __syncthreads();
  float d0 = v0 - mean, d1 = v1 - mean;
  red[tid] = d0 * d0 + d1 * d1; __syncthreads();
  for (int off = 128; off > 0; off >>= 1) { if (tid < off) red[tid] += red[tid + off]; __syncthreads(); }
  float rstd = rsqrtf(red[0] * (1.f / D_) + 1e-5f);
  float* yr = y + (size_t)row * D_;
  yr[tid]       = d0 * rstd * g[tid] + bb[tid];
  yr[tid + 256] = d1 * rstd * g[tid + 256] + bb[tid + 256];
}

// ---------------------------------------------------------------------------
// Host-side launch sequence
// ---------------------------------------------------------------------------
extern "C" void kernel_launch(void* const* d_in, const int* in_sizes, int n_in,
                              void* d_out, int out_size, void* d_ws, size_t ws_size,
                              hipStream_t stream) {
  (void)in_sizes; (void)n_in; (void)out_size; (void)ws_size;

  // ---- scratch arena ----
  char* w = (char*)d_ws; size_t off = 0;
  auto arena = [&](size_t bytes) { char* p = w + off; off += (bytes + 255) & ~(size_t)255; return p; };
  _Float16* xs16  = (_Float16*)arena((size_t)BLD * 2);
  _Float16* xt16  = (_Float16*)arena((size_t)BLD * 2);
  _Float16* a16   = (_Float16*)arena((size_t)BL * DFF * 2);   // staged A operand / f16 hidden
  _Float16* w16   = (_Float16*)arena((size_t)DFF * D_ * 2);   // staged weight
  float* Qf   = (float*)arena((size_t)BLD * 4);
  float* Kf   = (float*)arena((size_t)BLD * 4);
  float* Vf   = (float*)arena((size_t)BLD * 4);
  float* corr = (float*)arena((size_t)BLD * 4);
  float* mv   = (float*)arena((size_t)B_ * L_ * 4);
  int*   idxb = (int*)arena(64);
  float* smap = (float*)arena((size_t)B_ * 16 * 4);
  float* wsum = (float*)arena((size_t)B_ * H_ * L_ * 4);
  float* qsb  = (float*)arena((size_t)3 * B_ * H_ * E_ * 4);
  float* csb  = (float*)arena((size_t)3 * B_ * H_ * L_ * 4);
  float* xse  = (float*)arena((size_t)BLD * 4);
  float* xte  = (float*)arena((size_t)BLD * 4);
  float* xen  = (float*)arena((size_t)BLD * 4);
  float* trin = (float*)arena((size_t)BLD * 4);
  float* sein = (float*)arena((size_t)BLD * 4);
  float* tbuf = (float*)arena((size_t)BLD * 4);
  float* srec = (float*)arena((size_t)BLD * 4);
  float* din  = (float*)arena((size_t)BLD * 4);
  float* h1   = (float*)arena((size_t)BL * DFF * 4);          // also Qtr/Ktr staging
  float* m1   = (float*)arena((size_t)BLD * 4);
  _Float16* rt16 = (_Float16*)arena((size_t)B_ * F_ * D_ * 2);
  float* qr   = (float*)arena((size_t)B_ * F_ * D_ * 4);

  const float* xs = (const float*)d_in[0];
  const float* xt = (const float*)d_in[1];
  // param base indices (dict flatten order): sa=2, ta=8, sit=14, sit.trend=20, ts=26, sr=32
  const float* router = (const float*)d_in[38];
  const float* ln1g = (const float*)d_in[39]; const float* ln1b = (const float*)d_in[40];
  const float* ln2g = (const float*)d_in[41]; const float* ln2b = (const float*)d_in[42];
  const float* ln3g = (const float*)d_in[43]; const float* ln3b = (const float*)d_in[44];
  const float* ln4g = (const float*)d_in[45]; const float* ln4b = (const float*)d_in[46];

  auto cvt = [&](const float* src, _Float16* dst, int n) {
    f32_to_f16_kernel<<<(n + 255) / 256, 256, 0, stream>>>(src, dst, n);
  };
  auto gemm = [&](const _Float16* A, const _Float16* Bw, const float* bias, const float* addp,
                  float* C, int M, int N, int K, int relu) {
    dim3 g(N / 64, M / 64);
    gemm_f16_wmma<<<g, 256, 0, stream>>>(A, Bw, bias, addp, C, M, N, K, relu);
  };
  auto gemm_tdm = [&](const _Float16* A, const _Float16* Bw, const float* bias, const float* addp,
                      float* C, int M, int N, int K, int relu) {
    dim3 g(N / 64, M / 64);
    gemm_f16_wmma_tdm<<<g, 256, 0, stream>>>(A, Bw, bias, addp, C, M, N, K, relu);
  };
  auto proj = [&](const _Float16* A, int M, int widx, float* out) {
    cvt((const float*)d_in[widx], w16, D_ * D_);
    gemm(A, w16, (const float*)d_in[widx + 1], nullptr, out, M, D_, D_, 0);
  };
  auto season = [&](const float* Qp, const float* Kp) {
    season_corr_kernel<<<dim3(E_, H_, B_), 256, 0, stream>>>(Qp, Kp, corr);
    corr_mean_kernel<<<(B_ * L_ + 255) / 256, 256, 0, stream>>>(corr, mv);
    topk_softmax_kernel<<<1, 64, 0, stream>>>(mv, idxb, smap);
  };
  auto trend = [&](const float* Qp, int LqFull, const float* Kp, const float* Vp, float* outp) {
    zero_f32_kernel<<<(B_ * H_ * L_ + 255) / 256, 256, 0, stream>>>(wsum, B_ * H_ * L_);
    for (int i = 0; i < 3; ++i) {
      int st = i + 1, Lq = (LqFull + st - 1) / st;
      trend_attn_colsum<<<dim3(Lq, H_, B_), 256, 0, stream>>>(Qp, Kp, wsum, st, LqFull);
    }
    trend_apply_kernel<<<BLD / 256, 256, 0, stream>>>(Vp, wsum, outp, 1.f / 3.f);
  };

  cvt(xs, xs16, BLD);
  cvt(xt, xt16, BLD);

  // ---- S1: 'sa' season attention -> x_s_enc ----
  proj(xs16, BL, 2, Qf); proj(xs16, BL, 4, Kf); proj(xs16, BL, 6, Vf);
  season(Qf, Kf);
  delays_kernel<<<BLD / 256, 256, 0, stream>>>(Vf, idxb, smap, nullptr, xse);

  // ---- S2: 'ta' trend attention -> x_t_enc ----
  proj(xt16, BL, 8, Qf); proj(xt16, BL, 10, Kf); proj(xt16, BL, 12, Vf);
  trend(Qf, L_, Kf, Vf, xte);

  // ---- S3: 'sit' season attention with trend colsums -> x_enc ----
  proj(xs16, BL, 14, Qf); proj(xs16, BL, 16, Kf); proj(xs16, BL, 18, Vf);
  float* qtr = h1; float* ktr = h1 + BLD;
  proj(xt16, BL, 20, qtr); proj(xt16, BL, 22, ktr);     // trend V-proj not needed
  qsum_kernel<<<(3 * B_ * H_ * E_ + 255) / 256, 256, 0, stream>>>(qtr, qsb);
  colsum_kernel<<<(3 * B_ * H_ * L_ + 255) / 256, 256, 0, stream>>>(qsb, ktr, csb);
  season(Qf, Kf);
  delays_kernel<<<BLD / 256, 256, 0, stream>>>(Vf, idxb, smap, csb, xen);

  // ---- S4: layernorms (x + dropout(x) == 2x in eval) ----
  layernorm_kernel<<<BL, 256, 0, stream>>>(xte, ln1g, ln1b, trin, 2.0f);
  layernorm_kernel<<<BL, 256, 0, stream>>>(xse, ln2g, ln2b, sein, 2.0f);

  // ---- S5: 'ts' trend layer (router queries) -> trend_buffer ----
  router_rep_kernel<<<(B_ * F_ * D_ + 255) / 256, 256, 0, stream>>>(router, rt16);
  cvt((const float*)d_in[26], w16, D_ * D_);
  gemm(rt16, w16, (const float*)d_in[27], nullptr, qr, B_ * F_, D_, D_, 0);
  cvt(trin, a16, BLD);
  proj(a16, BL, 28, Kf);
  proj(a16, BL, 30, Vf);
  trend(qr, F_, Kf, Vf, tbuf);

  // ---- S6: 'sr' season attention (q=season_in, k=v=trend_buffer) ----
  cvt(sein, a16, BLD);
  proj(a16, BL, 32, Qf);
  cvt(tbuf, a16, BLD);
  proj(a16, BL, 34, Kf);
  proj(a16, BL, 36, Vf);
  season(Qf, Kf);
  delays_kernel<<<BLD / 256, 256, 0, stream>>>(Vf, idxb, smap, nullptr, srec);

  // ---- S7: MLPs + layernorms + residuals (TDM-fed WMMA GEMMs) ----
  layernorm_kernel<<<BL, 256, 0, stream>>>(xen, ln3g, ln3b, din, 2.0f);
  cvt(din, a16, BLD);
  cvt((const float*)d_in[47], w16, DFF * D_);
  gemm_tdm(a16, w16, (const float*)d_in[48], nullptr, h1, BL, DFF, D_, 1);
  cvt(h1, a16, BL * DFF);
  cvt((const float*)d_in[49], w16, D_ * DFF);
  gemm_tdm(a16, w16, (const float*)d_in[50], nullptr, m1, BL, D_, DFF, 0);
  add_kernel<<<BLD / 256, 256, 0, stream>>>(din, m1, din, BLD);
  layernorm_kernel<<<BL, 256, 0, stream>>>(din, ln4g, ln4b, m1, 1.0f);  // m1 = LN4

  cvt(srec, a16, BLD);
  cvt((const float*)d_in[51], w16, DFF * D_);
  gemm_tdm(a16, w16, (const float*)d_in[52], nullptr, h1, BL, DFF, D_, 1);
  cvt(h1, a16, BL * DFF);
  cvt((const float*)d_in[53], w16, D_ * DFF);
  gemm_tdm(a16, w16, (const float*)d_in[54], m1, (float*)d_out, BL, D_, DFF, 0);
}
